// ChannelBlock_31336081391781
// MI455X (gfx1250) — compile-verified, matrix operand order checked by reading, not run on hardware
//
#include <hip/hip_runtime.h>
#include <hip/hip_bf16.h>
#include <math.h>

typedef __attribute__((ext_vector_type(16))) _Float16 v16h;
typedef __attribute__((ext_vector_type(8)))  _Float16 v8h;
typedef __attribute__((ext_vector_type(8)))  float    v8f;
typedef int gv4i __attribute__((vector_size(16)));   // matches builtin param pointee

#define BDIM   8
#define HDIM   64
#define WDIM   64
#define NTOK   4096          // H*W
#define CDIM   384
#define CH2    768           // 2*C
#define CHID   1536          // 4*C
#define NHEAD  8
#define DHEAD  48
#define MROWS  32768         // B*N

// gfx1250 async global->LDS path (ASYNCcnt), guarded so fallback always compiles
#if defined(__HIP_DEVICE_COMPILE__) && \
    __has_builtin(__builtin_amdgcn_global_load_async_to_lds_b128) && \
    __has_builtin(__builtin_amdgcn_s_wait_asynccnt)
#define ASYNC_LDS 1
#else
#define ASYNC_LDS 0
#endif

union HV16 { v16h v; v8h h[2]; };

__device__ __forceinline__ float gelu_exact(float v) {
    return 0.5f * v * (1.0f + erff(v * 0.70710678118654752f));
}

__device__ __forceinline__ v8f wmma_f16(v16h a, v16h b, v8f c) {
    return __builtin_amdgcn_wmma_f32_16x16x32_f16(
        false, a, false, b, (short)0, c, false, false);
}

// copy 16 bytes global -> LDS (async on gfx1250, plain vector copy otherwise)
__device__ __forceinline__ void cp_async16(_Float16* l, const _Float16* g) {
#if ASYNC_LDS
    __builtin_amdgcn_global_load_async_to_lds_b128(
        (__attribute__((address_space(1))) gv4i*)g,
        (__attribute__((address_space(3))) gv4i*)l, 0, 0);
#else
    *(v8h*)l = *(const v8h*)g;
#endif
}

__device__ __forceinline__ void wait_async() {
#if ASYNC_LDS
    __builtin_amdgcn_s_wait_asynccnt(0);
#endif
}

// ---------------------------------------------------------------------------
// Weight convert: fp32 W[K][N] -> f16 Wt[N][K]
// ---------------------------------------------------------------------------
__global__ void wconv_kernel(const float* __restrict__ w, _Float16* __restrict__ wt,
                             int K, int N) {
    int idx = blockIdx.x * blockDim.x + threadIdx.x;
    if (idx >= K * N) return;
    int n = idx / K, k = idx % K;
    wt[idx] = (_Float16)w[(size_t)k * N + n];
}

// ---------------------------------------------------------------------------
// Depthwise 3x3 conv positional encoding + residual (fp32)
// ---------------------------------------------------------------------------
__global__ __launch_bounds__(256)
void dwconv_kernel(const float* __restrict__ in, const float* __restrict__ w,
                   const float* __restrict__ bias, float* __restrict__ out) {
    int idx = blockIdx.x * 256 + threadIdx.x;
    if (idx >= BDIM * NTOK * CDIM) return;
    int c = idx % CDIM;
    int n = (idx / CDIM) & (NTOK - 1);
    int b = idx / (CDIM * NTOK);
    int y = n >> 6, x = n & 63;
    float acc = bias[c];
    #pragma unroll
    for (int dy = -1; dy <= 1; dy++) {
        int yy = y + dy;
        if (yy < 0 || yy >= HDIM) continue;
        #pragma unroll
        for (int dx = -1; dx <= 1; dx++) {
            int xx = x + dx;
            if (xx < 0 || xx >= WDIM) continue;
            acc += in[((size_t)b * NTOK + yy * WDIM + xx) * CDIM + c] *
                   w[c * 9 + (dy + 1) * 3 + (dx + 1)];
        }
    }
    out[idx] = in[idx] + acc;
}

// ---------------------------------------------------------------------------
// LayerNorm over C=384, one wave per row (12 elems/lane), optional fp32 copy
// ---------------------------------------------------------------------------
template<bool W32>
__global__ __launch_bounds__(256)
void layernorm_kernel(const float* __restrict__ in, const float* __restrict__ g,
                      const float* __restrict__ bta, float* __restrict__ out32,
                      _Float16* __restrict__ out16) {
    int row  = blockIdx.x * 8 + (threadIdx.x >> 5);
    int lane = threadIdx.x & 31;
    const float* x = in + (size_t)row * CDIM;
    float vals[12];
    float s = 0.f;
    #pragma unroll
    for (int i = 0; i < 12; i++) { vals[i] = x[lane + i * 32]; s += vals[i]; }
    #pragma unroll
    for (int o = 16; o > 0; o >>= 1) s += __shfl_xor(s, o, 32);
    float mu = s * (1.0f / CDIM);
    float v = 0.f;
    #pragma unroll
    for (int i = 0; i < 12; i++) { float d = vals[i] - mu; v += d * d; }
    #pragma unroll
    for (int o = 16; o > 0; o >>= 1) v += __shfl_xor(v, o, 32);
    float inv = rsqrtf(v * (1.0f / CDIM) + 1e-5f);
    #pragma unroll
    for (int i = 0; i < 12; i++) {
        int c = lane + i * 32;
        float y = (vals[i] - mu) * inv * g[c] + bta[c];
        if (W32) out32[(size_t)row * CDIM + c] = y;
        out16[(size_t)row * CDIM + c] = (_Float16)y;
    }
}

// ---------------------------------------------------------------------------
// Tiled WMMA GEMM: out[M,N] = act(A[M,K](f16) @ Wt[N,K]^T(f16) + bias) (+res)
// 128x128 block tile, 8 waves of 32x64, K-step 32, double-buffered LDS with
// async global->LDS copies (ASYNCcnt) and one barrier per K-step.
// ---------------------------------------------------------------------------
template<int ACT, bool BIAS, bool RESID, bool OUT32>
__global__ __launch_bounds__(256)
void gemm_wmma_kernel(const _Float16* __restrict__ A, const _Float16* __restrict__ Wt,
                      const float* __restrict__ bias, const float* __restrict__ resid,
                      float* __restrict__ out32, _Float16* __restrict__ out16,
                      int M, int N, int K) {
    __shared__ _Float16 As[2][128 * 40];
    __shared__ _Float16 Bs[2][128 * 40];
    const int tid  = threadIdx.x;
    const int lane = tid & 31, wave = tid >> 5;
    const int wy = wave >> 1, wx = wave & 1;
    const int lr = lane & 15, lh = lane >> 4;
    const int m0 = blockIdx.y * 128, n0 = blockIdx.x * 128;
    const int lrow = tid >> 1, lseg = tid & 1;   // cooperative tile loads

    auto issue_tile = [&](int buf, int k0) {
        const _Float16* ga = A + (size_t)(m0 + lrow) * K + k0 + lseg * 16;
        _Float16* la = &As[buf][lrow * 40 + lseg * 16];
        cp_async16(la,     ga);
        cp_async16(la + 8, ga + 8);
        const _Float16* gb = Wt + (size_t)(n0 + lrow) * K + k0 + lseg * 16;
        _Float16* lb = &Bs[buf][lrow * 40 + lseg * 16];
        cp_async16(lb,     gb);
        cp_async16(lb + 8, gb + 8);
    };

    v8f acc[2][4];
    #pragma unroll
    for (int i = 0; i < 2; i++)
        #pragma unroll
        for (int j = 0; j < 4; j++) acc[i][j] = (v8f){};

    issue_tile(0, 0);
    int cur = 0;
    for (int k0 = 0; k0 < K; k0 += 32) {
        wait_async();          // my async slab complete
        __syncthreads();       // everyone's slab complete; prev reads of alt done
        if (k0 + 32 < K) issue_tile(cur ^ 1, k0 + 32);
        if (k0 + 64 < K) {     // global_prefetch_b8 of the K+64 slab
            __builtin_prefetch(A  + (size_t)(m0 + lrow) * K + k0 + 64 + lseg * 16, 0, 1);
            __builtin_prefetch(Wt + (size_t)(n0 + lrow) * K + k0 + 64 + lseg * 16, 0, 1);
        }

        v16h afr[2], bfr[4];
        #pragma unroll
        for (int ti = 0; ti < 2; ti++) {
            int r = wy * 32 + ti * 16 + lr;
            HV16 u;
            u.h[0] = *(const v8h*)(As[cur] + r * 40 + lh * 8);
            u.h[1] = *(const v8h*)(As[cur] + r * 40 + 16 + lh * 8);
            afr[ti] = u.v;
        }
        #pragma unroll
        for (int tj = 0; tj < 4; tj++) {
            int cidx = wx * 64 + tj * 16 + lr;
            HV16 u;
            u.h[0] = *(const v8h*)(Bs[cur] + cidx * 40 + lh * 8);
            u.h[1] = *(const v8h*)(Bs[cur] + cidx * 40 + 16 + lh * 8);
            bfr[tj] = u.v;
        }
        #pragma unroll
        for (int ti = 0; ti < 2; ti++)
            #pragma unroll
            for (int tj = 0; tj < 4; tj++)
                acc[ti][tj] = wmma_f16(afr[ti], bfr[tj], acc[ti][tj]);
        cur ^= 1;
    }

    // epilogue: VGPR r of C/D tile -> (M = base + r + lh*8, N = base + lr)
    #pragma unroll
    for (int ti = 0; ti < 2; ti++) {
        #pragma unroll
        for (int tj = 0; tj < 4; tj++) {
            int gmb = m0 + wy * 32 + ti * 16;
            int gn  = n0 + wx * 64 + tj * 16 + lr;
            float bv = BIAS ? bias[gn] : 0.f;
            #pragma unroll
            for (int r = 0; r < 8; r++) {
                int gm = gmb + r + lh * 8;
                float v = acc[ti][tj][r] + bv;
                if (ACT == 1) v = gelu_exact(v);
                if (RESID) v += resid[(size_t)gm * N + gn];
                if (OUT32) out32[(size_t)gm * N + gn] = v;
                else       out16[(size_t)gm * N + gn] = (_Float16)v;
            }
        }
    }
}

// ---------------------------------------------------------------------------
// Channel attention: logits[b,h,d,e] = sum_n (k[n,d]*scale) * v[n,e]
// One block per (b,h); 9 waves, one 16x16 tile each. k/v staged TRANSPOSED in
// LDS ([c][n], 64B rows) so fragments are two aligned ds_load_b128 per matrix.
// ---------------------------------------------------------------------------
__global__ __launch_bounds__(288)
void attn_kv_kernel(const _Float16* __restrict__ kv, float* __restrict__ logits) {
    const int bh = blockIdx.x;               // 0..63
    const int b = bh >> 3, h = bh & 7;
    __shared__ _Float16 kst[DHEAD * 32];     // [c][n]
    __shared__ _Float16 vst[DHEAD * 32];     // [c][n]
    const int tid = threadIdx.x, lane = tid & 31, wave = tid >> 5;
    const int ti = wave / 3, tj = wave % 3;
    const int lr = lane & 15, lh = lane >> 4;
    const float scale = 0.14433756729740643f;   // 48^-0.5
    v8f acc = (v8f){};
    for (int n0 = 0; n0 < NTOK; n0 += 32) {
        __syncthreads();
        for (int i = tid; i < 32 * DHEAD; i += 288) {
            int n = i / DHEAD, c = i % DHEAD;
            size_t base = ((size_t)(b * NTOK + n0 + n)) * CH2 + h * DHEAD + c;
            kst[c * 32 + n] = (_Float16)((float)kv[base] * scale);
            vst[c * 32 + n] = kv[base + CDIM];
        }
        __syncthreads();
        HV16 ua, ub;
        const int arow = (ti * 16 + lr) * 32;
        ua.h[0] = *(const v8h*)(kst + arow + lh * 8);        // A[M=d][K=n]
        ua.h[1] = *(const v8h*)(kst + arow + 16 + lh * 8);
        const int brow = (tj * 16 + lr) * 32;
        ub.h[0] = *(const v8h*)(vst + brow + lh * 8);        // B[K=n][N=e]
        ub.h[1] = *(const v8h*)(vst + brow + 16 + lh * 8);
        acc = wmma_f16(ua.v, ub.v, acc);
    }
    int cb = tj * 16 + lr;
    #pragma unroll
    for (int r = 0; r < 8; r++) {
        int rr = ti * 16 + r + lh * 8;
        logits[(size_t)bh * (DHEAD * DHEAD) + rr * DHEAD + cb] = acc[r];
    }
}

// ---------------------------------------------------------------------------
// Row softmax over 48, output f16 zero-padded to 64 columns (for K=64 WMMA)
// ---------------------------------------------------------------------------
__global__ void softmax48_kernel(const float* __restrict__ logits,
                                 _Float16* __restrict__ attnp) {
    int idx = blockIdx.x * blockDim.x + threadIdx.x;    // row = bh*48 + d
    if (idx >= 64 * DHEAD) return;
    const float* row = logits + (size_t)idx * DHEAD;
    float m = -3.4e38f;
    for (int j = 0; j < DHEAD; j++) m = fmaxf(m, row[j]);
    float s = 0.f;
    for (int j = 0; j < DHEAD; j++) s += expf(row[j] - m);
    float inv = 1.0f / s;
    _Float16* orow = attnp + (size_t)idx * 64;
    for (int j = 0; j < DHEAD; j++) orow[j] = (_Float16)(expf(row[j] - m) * inv);
    for (int j = DHEAD; j < 64; j++) orow[j] = (_Float16)0.f;
}

// ---------------------------------------------------------------------------
// out[b,h,n,d] = sum_e attn[d,e] * q[n,e]  (per (b,h):  q[N,48] @ attn^T)
// Block: 4 waves, 64 rows of N x 48 cols, K = 64 (zero-padded e).
// ---------------------------------------------------------------------------
__global__ __launch_bounds__(128)
void attn_apply_kernel(const _Float16* __restrict__ q, const _Float16* __restrict__ attnp,
                       _Float16* __restrict__ aoh) {
    const int bh = blockIdx.y;
    const int b = bh >> 3, h = bh & 7;
    const int n0 = blockIdx.x * 64;
    __shared__ _Float16 qs[64 * 64];
    __shared__ _Float16 as_[DHEAD * 64];
    const int tid = threadIdx.x, lane = tid & 31, wave = tid >> 5;
    const int lr = lane & 15, lh = lane >> 4;
    // zero-pad q columns 48..63 (plain stores)
    for (int i = tid; i < 64 * 16; i += 128) {
        int r = i >> 4, c = 48 + (i & 15);
        qs[r * 64 + c] = (_Float16)0.f;
    }
    // async 16B copies: q rows (6 x 16B each) and attn table (contig 16B)
    for (int i = tid; i < 64 * 6; i += 128) {
        int r = i / 6, ch = i % 6;
        cp_async16(qs + r * 64 + ch * 8,
                   q + ((size_t)(b * NTOK + n0 + r)) * CDIM + h * DHEAD + ch * 8);
    }
    for (int i = tid; i < (DHEAD * 64) / 8; i += 128)
        cp_async16(as_ + (size_t)i * 8, attnp + (size_t)bh * DHEAD * 64 + (size_t)i * 8);
    wait_async();
    __syncthreads();
    #pragma unroll
    for (int tj = 0; tj < 3; tj++) {
        v8f acc = (v8f){};
        #pragma unroll
        for (int kc = 0; kc < 64; kc += 32) {
            HV16 ua, ub;
            ua.h[0] = *(const v8h*)(qs  + (wave * 16 + lr) * 64 + kc + lh * 8);
            ua.h[1] = *(const v8h*)(qs  + (wave * 16 + lr) * 64 + kc + 16 + lh * 8);
            ub.h[0] = *(const v8h*)(as_ + (tj * 16 + lr) * 64 + kc + lh * 8);
            ub.h[1] = *(const v8h*)(as_ + (tj * 16 + lr) * 64 + kc + 16 + lh * 8);
            acc = wmma_f16(ua.v, ub.v, acc);
        }
        #pragma unroll
        for (int r = 0; r < 8; r++) {
            int gm = n0 + wave * 16 + r + lh * 8;
            aoh[((size_t)(b * NTOK + gm)) * CDIM + h * DHEAD + tj * 16 + lr] =
                (_Float16)acc[r];
        }
    }
}

// ---------------------------------------------------------------------------
extern "C" void kernel_launch(void* const* d_in, const int* in_sizes, int n_in,
                              void* d_out, int out_size, void* d_ws, size_t ws_size,
                              hipStream_t stream) {
    const float* x      = (const float*)d_in[0];
    const float* src    = (const float*)d_in[1];
    const float* cpe0_w = (const float*)d_in[4];
    const float* cpe0_b = (const float*)d_in[5];
    const float* cpe1_w = (const float*)d_in[6];
    const float* cpe1_b = (const float*)d_in[7];
    const float* n1g    = (const float*)d_in[8];
    const float* n1b    = (const float*)d_in[9];
    const float* q_w    = (const float*)d_in[10];
    const float* kv_w   = (const float*)d_in[11];
    const float* proj_w = (const float*)d_in[12];
    const float* proj_b = (const float*)d_in[13];
    const float* n2g    = (const float*)d_in[14];
    const float* n2b    = (const float*)d_in[15];
    const float* fc1_w  = (const float*)d_in[16];
    const float* fc1_b  = (const float*)d_in[17];
    const float* fc2_w  = (const float*)d_in[18];
    const float* fc2_b  = (const float*)d_in[19];
    float* out = (float*)d_out;

    // ---- workspace layout (all sizes multiples of 256 bytes) ----
    char* ws = (char*)d_ws;
    size_t o = 0;
    auto alloc = [&](size_t bytes) { size_t r = o; o += (bytes + 255) & ~(size_t)255; return r; };
    const size_t SZ_F32 = (size_t)MROWS * CDIM * 4;   // 50331648
    const size_t SZ_F16 = (size_t)MROWS * CDIM * 2;   // 25165824
    size_t o_xa    = alloc(SZ_F32);
    size_t o_sa    = alloc(SZ_F32);
    size_t o_xn32  = alloc(SZ_F32);        // | contiguous 100 MB region
    size_t o_xnh   = alloc(SZ_F16);        // | reused later as h1 (fc1 out)
    size_t o_snh   = alloc(SZ_F16);        // |
    size_t o_qh    = alloc(SZ_F16);        // reused later as h16 (LN2 out)
    size_t o_kvh   = alloc((size_t)MROWS * CH2 * 2);
    size_t o_aoh   = alloc(SZ_F16);
    size_t o_logit = alloc((size_t)64 * DHEAD * DHEAD * 4);
    size_t o_attnp = alloc((size_t)64 * DHEAD * 64 * 2);
    size_t o_qWt   = alloc((size_t)CDIM * CDIM * 2);
    size_t o_kvWt  = alloc((size_t)CDIM * CH2 * 2);
    size_t o_pWt   = alloc((size_t)CDIM * CDIM * 2);
    size_t o_f1Wt  = alloc((size_t)CDIM * CHID * 2);
    size_t o_f2Wt  = alloc((size_t)CHID * CDIM * 2);
    (void)ws_size; (void)in_sizes; (void)n_in; (void)out_size;

    float*     xa    = (float*)(ws + o_xa);
    float*     sa    = (float*)(ws + o_sa);
    float*     xn32  = (float*)(ws + o_xn32);
    _Float16*  xnh   = (_Float16*)(ws + o_xnh);
    _Float16*  snh   = (_Float16*)(ws + o_snh);
    _Float16*  qh    = (_Float16*)(ws + o_qh);
    _Float16*  kvh   = (_Float16*)(ws + o_kvh);
    _Float16*  aoh   = (_Float16*)(ws + o_aoh);
    float*     logit = (float*)(ws + o_logit);
    _Float16*  attnp = (_Float16*)(ws + o_attnp);
    _Float16*  qWt   = (_Float16*)(ws + o_qWt);
    _Float16*  kvWt  = (_Float16*)(ws + o_kvWt);
    _Float16*  pWt   = (_Float16*)(ws + o_pWt);
    _Float16*  f1Wt  = (_Float16*)(ws + o_f1Wt);
    _Float16*  f2Wt  = (_Float16*)(ws + o_f2Wt);
    // aliases over dead buffers
    float*     x2  = xa;                       // x after attn residual
    float*     x3  = sa;                       // x after cpe1
    _Float16*  h16 = qh;                       // LN2 output (f16)
    _Float16*  h1  = (_Float16*)(ws + o_xn32); // fc1 output [M, 1536] f16 (100 MB)

    const int elems = BDIM * NTOK * CDIM;
    const int cvb = 256;
    // 1) weight conversion (fp32 [K][N] -> f16 [N][K])
    wconv_kernel<<<(CDIM * CDIM + cvb - 1) / cvb, cvb, 0, stream>>>(q_w,    qWt,  CDIM, CDIM);
    wconv_kernel<<<(CDIM * CH2  + cvb - 1) / cvb, cvb, 0, stream>>>(kv_w,   kvWt, CDIM, CH2);
    wconv_kernel<<<(CDIM * CDIM + cvb - 1) / cvb, cvb, 0, stream>>>(proj_w, pWt,  CDIM, CDIM);
    wconv_kernel<<<(CDIM * CHID + cvb - 1) / cvb, cvb, 0, stream>>>(fc1_w,  f1Wt, CDIM, CHID);
    wconv_kernel<<<(CHID * CDIM + cvb - 1) / cvb, cvb, 0, stream>>>(fc2_w,  f2Wt, CHID, CDIM);

    // 2) CPE0 on x and source
    dwconv_kernel<<<(elems + 255) / 256, 256, 0, stream>>>(x,   cpe0_w, cpe0_b, xa);
    dwconv_kernel<<<(elems + 255) / 256, 256, 0, stream>>>(src, cpe0_w, cpe0_b, sa);

    // 3) LayerNorm1 (x keeps fp32 copy for residual; source f16-only)
    layernorm_kernel<true ><<<MROWS / 8, 256, 0, stream>>>(xa, n1g, n1b, xn32, xnh);
    layernorm_kernel<false><<<MROWS / 8, 256, 0, stream>>>(sa, n1g, n1b, nullptr, snh);

    // 4) q / kv projections (WMMA)
    gemm_wmma_kernel<0, false, false, false><<<dim3(CDIM / 128, MROWS / 128), 256, 0, stream>>>(
        xnh, qWt, nullptr, nullptr, nullptr, qh, MROWS, CDIM, CDIM);
    gemm_wmma_kernel<0, false, false, false><<<dim3(CH2 / 128, MROWS / 128), 256, 0, stream>>>(
        snh, kvWt, nullptr, nullptr, nullptr, kvh, MROWS, CH2, CDIM);

    // 5) channel cross-covariance + softmax
    attn_kv_kernel<<<64, 288, 0, stream>>>(kvh, logit);
    softmax48_kernel<<<(64 * DHEAD + 255) / 256, 256, 0, stream>>>(logit, attnp);

    // 6) apply attention (q @ attn^T per head, WMMA)
    attn_apply_kernel<<<dim3(NTOK / 64, 64), 128, 0, stream>>>(qh, attnp, aoh);

    // 7) proj + bias + residual on normalized x -> x2 (fp32)
    gemm_wmma_kernel<0, true, true, true><<<dim3(CDIM / 128, MROWS / 128), 256, 0, stream>>>(
        aoh, pWt, proj_b, xn32, x2, nullptr, MROWS, CDIM, CDIM);

    // 8) CPE1 -> x3
    dwconv_kernel<<<(elems + 255) / 256, 256, 0, stream>>>(x2, cpe1_w, cpe1_b, x3);

    // 9) LayerNorm2 -> h16
    layernorm_kernel<false><<<MROWS / 8, 256, 0, stream>>>(x3, n2g, n2b, nullptr, h16);

    // 10) MLP: fc1 + exact GELU -> h1 ; fc2 + bias + residual(x3) -> out
    gemm_wmma_kernel<1, true, false, false><<<dim3(CHID / 128, MROWS / 128), 256, 0, stream>>>(
        h16, f1Wt, fc1_b, nullptr, nullptr, h1, MROWS, CHID, CDIM);
    gemm_wmma_kernel<0, true, true, true><<<dim3(CDIM / 128, MROWS / 128), 256, 0, stream>>>(
        h1, f2Wt, fc2_b, x3, out, nullptr, MROWS, CDIM, CHID);
}
